// SimCSE_71940702208246
// MI455X (gfx1250) — matrix operand order repaired and checked by reading, not compiled, not run on hardware
//
#include <hip/hip_runtime.h>

typedef __attribute__((ext_vector_type(16))) __bf16 v16bf;
typedef __attribute__((ext_vector_type(8)))  float  v8f;

#define NROWS 8192
#define DDIM  768
#define LDA   776      // padded LDS row stride (elements): 16B pad -> conflict-free b128 reads
#define INV_TEMP 20.0f // 1 / 0.05
#define SHIFT 20.0f    // fixed logsumexp shift: |sim| <= 20, sum(exp(sim-20)) <= N, no overflow

// ---------------------------------------------------------------------------
// Kernel 1: row L2-normalize f32 -> bf16 (one 256-thread block per row)
// ---------------------------------------------------------------------------
__global__ __launch_bounds__(256) void normalize_rows(const float* __restrict__ in,
                                                      __bf16* __restrict__ out) {
  const int row = blockIdx.x;
  const float* src = in + (size_t)row * DDIM;
  __bf16* dst = out + (size_t)row * DDIM;

  float ss = 0.f;
  for (int i = threadIdx.x; i < DDIM; i += 256) { float v = src[i]; ss = fmaf(v, v, ss); }
  for (int m = 16; m >= 1; m >>= 1) ss += __shfl_xor(ss, m, 32);

  __shared__ float red[8];
  __shared__ float scale_s;
  if ((threadIdx.x & 31) == 0) red[threadIdx.x >> 5] = ss;
  __syncthreads();
  if (threadIdx.x == 0) {
    float t = 0.f;
#pragma unroll
    for (int w = 0; w < 8; ++w) t += red[w];
    scale_s = 1.0f / fmaxf(sqrtf(t), 1e-8f);
  }
  __syncthreads();
  const float scale = scale_s;
  for (int i = threadIdx.x; i < DDIM; i += 256) dst[i] = (__bf16)(src[i] * scale);
}

// ---------------------------------------------------------------------------
// Kernel 2: fused sim-tile GEMM (bf16 WMMA) + streaming logsumexp + diagonal.
// Block = 4 waves (128 thr). Block owns 16 rows; wave owns a 2048-col stripe.
// K loop is unrolled x2 with explicit double-buffered A/B fragments so B
// prefetches overlap WMMA issue (partial s_wait_loadcnt instead of ==0).
// ---------------------------------------------------------------------------
__global__ __launch_bounds__(128) void simcse_fused(const __bf16* __restrict__ Qn,
                                                    const __bf16* __restrict__ Cn,
                                                    float* __restrict__ rowOut) {
  __shared__ __align__(16) __bf16 Atile[16 * LDA];
  __shared__ float redS[4][16];
  __shared__ float redD[4][16];

  const int rowBase = blockIdx.x * 16;
  const int wave = threadIdx.x >> 5;
  const int lane = threadIdx.x & 31;
  const int half = lane >> 4;   // half-wave: selects K sub-block of the fragment
  const int lmod = lane & 15;   // row (A) / column (B,D) within the 16-wide tile

  // Stage the 16-row Q tile into LDS, 16B vectors, padded row stride.
  for (int c = threadIdx.x; c < 16 * (DDIM / 8); c += 128) {
    const int r  = c / (DDIM / 8);
    const int c8 = c % (DDIM / 8);
    uint4 v = ((const uint4*)(Qn + (size_t)(rowBase + r) * DDIM))[c8];
    *(uint4*)(&Atile[r * LDA + c8 * 8]) = v;
  }
  __syncthreads();

  float s[8];     // per-lane partial sum of exp(sim - SHIFT); rows half*8+r
  float dval[8];  // diagonal capture (exactly one lane/wave sees each row's diag)
#pragma unroll
  for (int r = 0; r < 8; ++r) { s[r] = 0.f; dval[r] = 0.f; }

  // Per-lane base pointers for fragment loads.
  const __bf16* aBase = &Atile[lmod * LDA + half * 8];
  const int colStart = wave * (NROWS / 4);

  // A fragment (16x32 bf16): lane holds row lmod, K = k+half*8..+7 and k+16+half*8..+7
  auto loadA = [&](v16bf& a, int k) {
    const __bf16* ap = aBase + k;
    ((uint4*)&a)[0] = *(const uint4*)(ap);
    ((uint4*)&a)[1] = *(const uint4*)(ap + 16);
  };

  for (int it = 0; it < NROWS / 4; it += 64) {
    const int nbase0 = colStart + it;
    // B fragment (32x16 bf16): B[kk][n] = Cn[n][kk] -> lane holds bf16 of row
    // (nbase0 + t*16 + lmod) at K = k+half*8..+7 and k+16+half*8..+7.
    const __bf16* bBase = Cn + (size_t)(nbase0 + lmod) * DDIM + half * 8;
    auto loadB = [&](v16bf* b, int k) {
#pragma unroll
      for (int t = 0; t < 4; ++t) {
        const __bf16* bp = bBase + (size_t)t * 16 * DDIM + k;
        ((uint4*)&b[t])[0] = ((const uint4*)bp)[0];
        ((uint4*)&b[t])[1] = ((const uint4*)bp)[2];
      }
    };

    v8f acc[4] = {v8f{}, v8f{}, v8f{}, v8f{}};
    v16bf a0, a1, b0[4], b1[4];

    loadB(b0, 0);
    loadA(a0, 0);
    for (int k = 0; k < DDIM; k += 64) {
      // Prefetch K+32 while consuming K.
      loadB(b1, k + 32);
      loadA(a1, k + 32);
#pragma unroll
      for (int t = 0; t < 4; ++t)
        acc[t] = __builtin_amdgcn_wmma_f32_16x16x32_bf16(
            false, a0, false, b0[t], (short)0, acc[t], false, false);

      // Prefetch K+64 (wraps to 0 on the last step: branch-free, redundant load).
      int knext = k + 64;
      knext = (knext < DDIM) ? knext : 0;
      loadB(b0, knext);
      loadA(a0, knext);
#pragma unroll
      for (int t = 0; t < 4; ++t)
        acc[t] = __builtin_amdgcn_wmma_f32_16x16x32_bf16(
            false, a1, false, b1[t], (short)0, acc[t], false, false);
    }

    // Streaming LSE accumulate (fixed shift; no cross-lane work needed here).
    // D layout: VGPR r, lanes 0-15: M=r, N=lane; lanes 16-31: M=r+8, N=lane-16.
#pragma unroll
    for (int t = 0; t < 4; ++t) {
      const int gcol = nbase0 + t * 16 + lmod;
#pragma unroll
      for (int r = 0; r < 8; ++r) {
        const float val = acc[t][r] * INV_TEMP;
        if (gcol == rowBase + half * 8 + r) dval[r] = val;
        s[r] += __expf(val - SHIFT);
      }
    }
  }

  // Reduce across the 16 lanes of each half-wave (row-replicated after this).
#pragma unroll
  for (int r = 0; r < 8; ++r) {
    float sv = s[r], dv = dval[r];
    for (int m = 8; m >= 1; m >>= 1) {
      sv += __shfl_xor(sv, m, 32);
      dv += __shfl_xor(dv, m, 32);
    }
    if (lmod == 0) { redS[wave][half * 8 + r] = sv; redD[wave][half * 8 + r] = dv; }
  }
  __syncthreads();

  // Combine the 4 waves' disjoint column stripes; emit per-row loss term.
  if (threadIdx.x < 16) {
    const int r = threadIdx.x;
    const float S  = redS[0][r] + redS[1][r] + redS[2][r] + redS[3][r];
    const float dg = redD[0][r] + redD[1][r] + redD[2][r] + redD[3][r];
    rowOut[rowBase + r] = (SHIFT + __logf(S)) - dg;
  }
}

// ---------------------------------------------------------------------------
// Kernel 3: deterministic mean of the 8192 per-row terms -> scalar loss
// ---------------------------------------------------------------------------
__global__ __launch_bounds__(256) void reduce_mean(const float* __restrict__ rowOut,
                                                   float* __restrict__ out) {
  __shared__ float sh[256];
  float acc = 0.f;
  for (int i = threadIdx.x; i < NROWS; i += 256) acc += rowOut[i];
  sh[threadIdx.x] = acc;
  __syncthreads();
  for (int sft = 128; sft > 0; sft >>= 1) {
    if (threadIdx.x < sft) sh[threadIdx.x] += sh[threadIdx.x + sft];
    __syncthreads();
  }
  if (threadIdx.x == 0) out[0] = sh[0] * (1.0f / (float)NROWS);
}

// ---------------------------------------------------------------------------
extern "C" void kernel_launch(void* const* d_in, const int* in_sizes, int n_in,
                              void* d_out, int out_size, void* d_ws, size_t ws_size,
                              hipStream_t stream) {
  (void)in_sizes; (void)n_in; (void)out_size; (void)ws_size;
  const float* Q = (const float*)d_in[0];
  const float* C = (const float*)d_in[1];

  // Workspace layout: Qn bf16 | Cn bf16 | rowOut f32   (~25.2 MB total)
  __bf16* Qn = (__bf16*)d_ws;
  __bf16* Cn = Qn + (size_t)NROWS * DDIM;
  float* rowOut = (float*)(Cn + (size_t)NROWS * DDIM);

  normalize_rows<<<NROWS, 256, 0, stream>>>(Q, Qn);
  normalize_rows<<<NROWS, 256, 0, stream>>>(C, Cn);
  simcse_fused<<<NROWS / 16, 128, 0, stream>>>(Qn, Cn, rowOut);
  reduce_mean<<<1, 256, 0, stream>>>(rowOut, (float*)d_out);
}
